// ExportableModel_1649267441697
// MI455X (gfx1250) — matrix operand order repaired and checked by reading, not compiled, not run on hardware
//
#include <hip/hip_runtime.h>
#include <math.h>

// ---------------- problem constants (from reference setup_inputs) ----------
#define N_NODES 20000
#define FIN     256
#define HDIM    512
#define H2      1024
#define E_TOT   150000
#define LEN0    75000
#define LEN1    75000
#define EPS_MSG 1e-7f
#define EPS_SM  1e-16f
#define BN_EPS  1e-5f

typedef __attribute__((ext_vector_type(16))) __bf16 v16bf;
typedef __attribute__((ext_vector_type(8)))  __bf16 v8bf;
typedef __attribute__((ext_vector_type(4)))  __bf16 v4bf;
typedef __attribute__((ext_vector_type(8)))  float  v8f;

__device__ inline v8f v8f_zero() {
  v8f z = {0.f,0.f,0.f,0.f,0.f,0.f,0.f,0.f};
  return z;
}

// ---------------- bf16 fragment loaders (pre-converted operands) -----------
// A fragment, 16x32 bf16 (M x K).  Lane L (row = m0 + (L&15)):
//   lanes 0-15 : elems 0..15 = K 0..15
//   lanes 16-31: elems 0..15 = K 16..31
// p = &A[row][kb + (hi?16:0)] ; reads p[0..15]  (2x b128)
__device__ inline v16bf ld_frag_a(const __bf16* __restrict__ p) {
  v8bf lo = *(const v8bf*)(p);
  v8bf hi = *(const v8bf*)(p + 8);
  return __builtin_shufflevector(lo, hi, 0,1,2,3,4,5,6,7,8,9,10,11,12,13,14,15);
}
// B fragment, 32x16 bf16 (K x N); B[k][n] = W[n][k], W row-major [Nc x K].
// Lane L (col = n0 + (L&15)): lanes 0-15 hold K 0..15, lanes 16-31 K 16..31.
// p = &W[col][kb + (hi?16:0)] ; reads p[0..15]  (2x b128)
__device__ inline v16bf ld_frag_b(const __bf16* __restrict__ p) {
  v8bf lo = *(const v8bf*)(p);
  v8bf hi = *(const v8bf*)(p + 8);
  return __builtin_shufflevector(lo, hi, 0,1,2,3,4,5,6,7,8,9,10,11,12,13,14,15);
}

// ---------------- f32 -> bf16 bulk convert (8 elems / thread) --------------
__global__ __launch_bounds__(256) void cvt_bf16_kernel(
    const float* __restrict__ in, __bf16* __restrict__ out, int n8)
{
  const int i = blockIdx.x * blockDim.x + threadIdx.x;
  if (i >= n8) return;
  float4 a = ((const float4*)in)[2 * i];
  float4 b = ((const float4*)in)[2 * i + 1];
  v8bf o;
  o[0]=(__bf16)a.x; o[1]=(__bf16)a.y; o[2]=(__bf16)a.z; o[3]=(__bf16)a.w;
  o[4]=(__bf16)b.x; o[5]=(__bf16)b.y; o[6]=(__bf16)b.z; o[7]=(__bf16)b.w;
  ((v8bf*)out)[i] = o;
}

// ---------------- GEMM:  C[M x Nc] = A[M x K] @ W[Nc x K]^T ----------------
// A, W pre-converted bf16.  One wave owns a 32x64 macro-tile (2x4 WMMA tiles).
// MODE 0: store f32          (sf / df)
// MODE 1: BN + ReLU -> bf16  (h, consumed by the W2 GEMM)
// MODE 2: f32 accumulate     (y += h @ W2^T)
template <int MODE>
__global__ __launch_bounds__(256) void gemm_wmma_kernel(
    const __bf16* __restrict__ A, const __bf16* __restrict__ W,
    float* __restrict__ Cf, __bf16* __restrict__ Cb,
    int M, int K, int Nc,
    const float* __restrict__ bn_g, const float* __restrict__ bn_b,
    const float* __restrict__ bn_m, const float* __restrict__ bn_v)
{
  const int lane = threadIdx.x & 31;
  const int wave = blockIdx.x * (blockDim.x >> 5) + (threadIdx.x >> 5);
  const int ntN  = Nc >> 6;                   // 64-wide N macro-tiles
  const int nWav = (M >> 5) * ntN;
  if (wave >= nWav) return;                   // wave-uniform: EXEC all-ones at WMMA
  const int m0 = (wave / ntN) << 5;
  const int n0 = (wave % ntN) << 6;

  const int l15 = lane & 15;
  const int hi  = (lane >> 4) & 1;

  const __bf16* pa0 = A + (size_t)(m0 + l15) * K + (hi ? 16 : 0);
  const __bf16* pa1 = pa0 + (size_t)16 * K;
  const __bf16* pb  = W + (size_t)(n0 + l15) * K + (hi ? 16 : 0);
  const size_t  wstep = (size_t)16 * K;

  v8f acc[2][4];
#pragma unroll
  for (int i = 0; i < 2; ++i)
#pragma unroll
    for (int j = 0; j < 4; ++j) acc[i][j] = v8f_zero();

  for (int kb = 0; kb < K; kb += 32) {
    v16bf a0 = ld_frag_a(pa0 + kb);
    v16bf a1 = ld_frag_a(pa1 + kb);
    v16bf b0 = ld_frag_b(pb + kb);
    v16bf b1 = ld_frag_b(pb + wstep + kb);
    v16bf b2 = ld_frag_b(pb + 2 * wstep + kb);
    v16bf b3 = ld_frag_b(pb + 3 * wstep + kb);
    acc[0][0] = __builtin_amdgcn_wmma_f32_16x16x32_bf16(false, a0, false, b0, (short)0, acc[0][0], false, false);
    acc[0][1] = __builtin_amdgcn_wmma_f32_16x16x32_bf16(false, a0, false, b1, (short)0, acc[0][1], false, false);
    acc[0][2] = __builtin_amdgcn_wmma_f32_16x16x32_bf16(false, a0, false, b2, (short)0, acc[0][2], false, false);
    acc[0][3] = __builtin_amdgcn_wmma_f32_16x16x32_bf16(false, a0, false, b3, (short)0, acc[0][3], false, false);
    acc[1][0] = __builtin_amdgcn_wmma_f32_16x16x32_bf16(false, a1, false, b0, (short)0, acc[1][0], false, false);
    acc[1][1] = __builtin_amdgcn_wmma_f32_16x16x32_bf16(false, a1, false, b1, (short)0, acc[1][1], false, false);
    acc[1][2] = __builtin_amdgcn_wmma_f32_16x16x32_bf16(false, a1, false, b2, (short)0, acc[1][2], false, false);
    acc[1][3] = __builtin_amdgcn_wmma_f32_16x16x32_bf16(false, a1, false, b3, (short)0, acc[1][3], false, false);
  }

  // C tile layout: VGPR r -> row m0 + 16*i + r + (hi?8:0); col = n0 + 16*j + (lane&15)
  const int rbase = m0 + hi * 8;
  int   col[4];
  float sc[4], of[4];
#pragma unroll
  for (int j = 0; j < 4; ++j) {
    col[j] = n0 + 16 * j + l15;
    if (MODE == 1) {
      const int c = col[j];
      sc[j] = bn_g[c] * rsqrtf(bn_v[c] + BN_EPS);
      of[j] = bn_b[c] - bn_m[c] * sc[j];
    }
  }

#pragma unroll
  for (int i = 0; i < 2; ++i) {
#pragma unroll
    for (int r = 0; r < 8; ++r) {
      const size_t rowoff = (size_t)(rbase + 16 * i + r) * Nc;
#pragma unroll
      for (int j = 0; j < 4; ++j) {
        const float z = acc[i][j][r];
        if (MODE == 0) {
          Cf[rowoff + col[j]] = z;
        } else if (MODE == 1) {
          Cb[rowoff + col[j]] = (__bf16)fmaxf(z * sc[j] + of[j], 0.f);
        } else {
          Cf[rowoff + col[j]] += z;
        }
      }
    }
  }
}

// ---------------- edge pass 1: segment max of msg into mx (int-bits trick) --
// msg = relu(sf[src] + ea*We) + EPS_MSG > 0, so int compare == float compare
// and memset-0 init acts as -inf for this positive domain.
// One wave per edge; lane covers 16 features as 4 strided float4 chunks.
__global__ __launch_bounds__(256) void edge_max_kernel(
    const float* __restrict__ sfeat, const int* __restrict__ src,
    const int* __restrict__ dst, const float* __restrict__ ea,
    const float* __restrict__ We, int* __restrict__ mx, int len)
{
  const int e = blockIdx.x * 8 + (threadIdx.x >> 5);
  if (e >= len) return;
  const int lane = threadIdx.x & 31;
  const int s = src[e], d = dst[e];
  const float a = ea[e];
  const float* srow = sfeat + (size_t)s * HDIM;
  int* mrow = mx + (size_t)d * HDIM;
#pragma unroll
  for (int c = 0; c < 4; ++c) {
    const int j = (lane + c * 32) * 4;
    float4 sv = *(const float4*)(srow + j);
    float4 wv = *(const float4*)(We + j);
    float m0 = fmaxf(fmaf(a, wv.x, sv.x), 0.f) + EPS_MSG;
    float m1 = fmaxf(fmaf(a, wv.y, sv.y), 0.f) + EPS_MSG;
    float m2 = fmaxf(fmaf(a, wv.z, sv.z), 0.f) + EPS_MSG;
    float m3 = fmaxf(fmaf(a, wv.w, sv.w), 0.f) + EPS_MSG;
    atomicMax(mrow + j + 0, __float_as_int(m0));
    atomicMax(mrow + j + 1, __float_as_int(m1));
    atomicMax(mrow + j + 2, __float_as_int(m2));
    atomicMax(mrow + j + 3, __float_as_int(m3));
  }
}

// ---------------- edge pass 2: den += ex, num += msg*ex ---------------------
__global__ __launch_bounds__(256) void edge_sum_kernel(
    const float* __restrict__ sfeat, const int* __restrict__ src,
    const int* __restrict__ dst, const float* __restrict__ ea,
    const float* __restrict__ We, const float* __restrict__ mx,
    float* __restrict__ den, float* __restrict__ num, int len)
{
  const int e = blockIdx.x * 8 + (threadIdx.x >> 5);
  if (e >= len) return;
  const int lane = threadIdx.x & 31;
  const int s = src[e], d = dst[e];
  const float a = ea[e];
  const float* srow = sfeat + (size_t)s * HDIM;
  const float* mrow = mx + (size_t)d * HDIM;
  float* drow = den + (size_t)d * HDIM;
  float* nrow = num + (size_t)d * HDIM;
#pragma unroll
  for (int c = 0; c < 4; ++c) {
    const int j = (lane + c * 32) * 4;
    float4 sv = *(const float4*)(srow + j);
    float4 wv = *(const float4*)(We + j);
    float4 mv = *(const float4*)(mrow + j);
    float m0 = fmaxf(fmaf(a, wv.x, sv.x), 0.f) + EPS_MSG;
    float m1 = fmaxf(fmaf(a, wv.y, sv.y), 0.f) + EPS_MSG;
    float m2 = fmaxf(fmaf(a, wv.z, sv.z), 0.f) + EPS_MSG;
    float m3 = fmaxf(fmaf(a, wv.w, sv.w), 0.f) + EPS_MSG;
    float e0 = __expf(m0 - mv.x);
    float e1 = __expf(m1 - mv.y);
    float e2 = __expf(m2 - mv.z);
    float e3 = __expf(m3 - mv.w);
    unsafeAtomicAdd(drow + j + 0, e0);
    unsafeAtomicAdd(drow + j + 1, e1);
    unsafeAtomicAdd(drow + j + 2, e2);
    unsafeAtomicAdd(drow + j + 3, e3);
    unsafeAtomicAdd(nrow + j + 0, m0 * e0);
    unsafeAtomicAdd(nrow + j + 1, m1 * e1);
    unsafeAtomicAdd(nrow + j + 2, m2 * e2);
    unsafeAtomicAdd(nrow + j + 3, m3 * e3);
  }
}

// ------- node combine: out = num/(den+eps) + df, emitted directly in bf16 ---
__global__ __launch_bounds__(256) void combine_kernel(
    const float* __restrict__ num, const float* __restrict__ den,
    const float* __restrict__ df, __bf16* __restrict__ outb, int n4)
{
  const int i = blockIdx.x * blockDim.x + threadIdx.x;
  if (i >= n4) return;
  float4 nv = ((const float4*)num)[i];
  float4 dv = ((const float4*)den)[i];
  float4 fv = ((const float4*)df)[i];
  v4bf o;
  o[0] = (__bf16)(nv.x / (dv.x + EPS_SM) + fv.x);
  o[1] = (__bf16)(nv.y / (dv.y + EPS_SM) + fv.y);
  o[2] = (__bf16)(nv.z / (dv.z + EPS_SM) + fv.z);
  o[3] = (__bf16)(nv.w / (dv.w + EPS_SM) + fv.w);
  ((v4bf*)outb)[i] = o;
}

// ---------------- leaky relu (slope 0.01, jax default) ----------------------
__global__ __launch_bounds__(256) void leaky_kernel(
    const float* __restrict__ y, float* __restrict__ x1, int n4)
{
  const int i = blockIdx.x * blockDim.x + threadIdx.x;
  if (i >= n4) return;
  float4 v = ((const float4*)y)[i];
  v.x = v.x > 0.f ? v.x : 0.01f * v.x;
  v.y = v.y > 0.f ? v.y : 0.01f * v.y;
  v.z = v.z > 0.f ? v.z : 0.01f * v.z;
  v.w = v.w > 0.f ? v.w : 0.01f * v.w;
  ((float4*)x1)[i] = v;
}

// ---------------- host-side glue -------------------------------------------
static inline void launch_gemm(int mode, const __bf16* A, const __bf16* W,
                               float* Cf, __bf16* Cb, int M, int K, int Nc,
                               const float* g, const float* b,
                               const float* m, const float* v, hipStream_t s)
{
  const int waves  = (M >> 5) * (Nc >> 6);
  const int blocks = (waves + 7) / 8;      // 8 waves (256 threads) per block
  switch (mode) {
    case 0: gemm_wmma_kernel<0><<<blocks, 256, 0, s>>>(A, W, Cf, Cb, M, K, Nc, g, b, m, v); break;
    case 1: gemm_wmma_kernel<1><<<blocks, 256, 0, s>>>(A, W, Cf, Cb, M, K, Nc, g, b, m, v); break;
    default: gemm_wmma_kernel<2><<<blocks, 256, 0, s>>>(A, W, Cf, Cb, M, K, Nc, g, b, m, v); break;
  }
}

static inline void launch_cvt(const float* in, __bf16* out, size_t n, hipStream_t s)
{
  const int n8 = (int)(n / 8);
  cvt_bf16_kernel<<<(n8 + 255) / 256, 256, 0, s>>>(in, out, n8);
}

extern "C" void kernel_launch(void* const* d_in, const int* in_sizes, int n_in,
                              void* d_out, int out_size, void* d_ws, size_t ws_size,
                              hipStream_t stream)
{
  const float* x       = (const float*)d_in[0];
  const int*   ei      = (const int*)  d_in[1];
  const float* ea      = (const float*)d_in[2];
  const float* l0_src  = (const float*)d_in[5];
  const float* l0_dst  = (const float*)d_in[6];
  const float* l0_edge = (const float*)d_in[7];
  const float* l0_w1   = (const float*)d_in[8];
  const float* l0_g    = (const float*)d_in[9];
  const float* l0_b    = (const float*)d_in[10];
  const float* l0_m    = (const float*)d_in[11];
  const float* l0_v    = (const float*)d_in[12];
  const float* l0_w2   = (const float*)d_in[13];
  const float* l1_edge = (const float*)d_in[14];
  const float* l1_w1   = (const float*)d_in[15];
  const float* l1_g    = (const float*)d_in[16];
  const float* l1_b    = (const float*)d_in[17];
  const float* l1_m    = (const float*)d_in[18];
  const float* l1_v    = (const float*)d_in[19];
  const float* l1_w2   = (const float*)d_in[20];

  const size_t NH   = (size_t)N_NODES * HDIM;   // 10.24M
  const size_t NFIN = (size_t)N_NODES * FIN;    // 5.12M
  const size_t W_SD = (size_t)2 * HDIM * FIN;   // 262144   (Ws or Wd, both links)
  const size_t W_12 = (size_t)2 * H2 * HDIM;    // 1048576  (W1 or W2, both links)

  // ---- workspace layout: f32 region then bf16 region ----
  float* y    = (float*)d_ws;        // NH
  float* x1   = y   + NH;            // NH
  float* mx   = x1  + NH;            // NH (float bits, int atomics)
  float* den  = mx  + NH;            // NH
  float* num  = den + NH;            // NH
  float* sf   = num + NH;            // NH
  float* df   = sf  + NH;            // NH
  __bf16* outb  = (__bf16*)(df + NH);      // NH       (GEMM-A for W1)
  __bf16* hb    = outb + NH;               // 2*NH     (GEMM-A for W2)
  __bf16* xb    = hb + 2 * NH;             // NFIN     (GEMM-A for Ws/Wd)
  __bf16* wsrcb = xb + NFIN;               // W_SD
  __bf16* wdstb = wsrcb + W_SD;            // W_SD
  __bf16* w1l0b = wdstb + W_SD;            // W_12
  __bf16* w2l0b = w1l0b + W_12;            // W_12
  __bf16* w1l1b = w2l0b + W_12;            // W_12
  __bf16* w2l1b = w1l1b + W_12;            // W_12

  // ---- one-time (per launch) bf16 staging of GEMM operands ----
  launch_cvt(x,      xb,    NFIN, stream);
  launch_cvt(l0_src, wsrcb, W_SD, stream);
  launch_cvt(l0_dst, wdstb, W_SD, stream);
  launch_cvt(l0_w1,  w1l0b, W_12, stream);
  launch_cvt(l0_w2,  w2l0b, W_12, stream);
  launch_cvt(l1_w1,  w1l1b, W_12, stream);
  launch_cvt(l1_w2,  w2l1b, W_12, stream);

  const int* srcAll = ei;
  const int* dstAll = ei + E_TOT;
  const int  lens[2] = {LEN0, LEN1};
  const int  offs[2] = {0, LEN0};

  const int n4      = (int)(NH / 4);
  const int ewBlock = (n4 + 255) / 256;

  // ---------------- layer 0 ----------------
  hipMemsetAsync(y, 0, NH * sizeof(float), stream);
  for (int l = 0; l < 2; ++l) {
    launch_gemm(0, xb, wsrcb + (size_t)l * HDIM * FIN, sf, nullptr,
                N_NODES, FIN, HDIM, nullptr, nullptr, nullptr, nullptr, stream);
    launch_gemm(0, xb, wdstb + (size_t)l * HDIM * FIN, df, nullptr,
                N_NODES, FIN, HDIM, nullptr, nullptr, nullptr, nullptr, stream);
    hipMemsetAsync(mx,  0, NH * sizeof(float), stream);
    hipMemsetAsync(den, 0, NH * sizeof(float), stream);
    hipMemsetAsync(num, 0, NH * sizeof(float), stream);
    const int*   srcL = srcAll + offs[l];
    const int*   dstL = dstAll + offs[l];
    const float* eaL  = ea + offs[l];
    const float* WeL  = l0_edge + (size_t)l * HDIM;
    const int eGrid = (lens[l] + 7) / 8;
    edge_max_kernel<<<eGrid, 256, 0, stream>>>(sf, srcL, dstL, eaL, WeL, (int*)mx, lens[l]);
    edge_sum_kernel<<<eGrid, 256, 0, stream>>>(sf, srcL, dstL, eaL, WeL, mx, den, num, lens[l]);
    combine_kernel<<<ewBlock, 256, 0, stream>>>(num, den, df, outb, n4);
    launch_gemm(1, outb, w1l0b + (size_t)l * H2 * HDIM, nullptr, hb,
                N_NODES, HDIM, H2,
                l0_g + (size_t)l * H2, l0_b + (size_t)l * H2,
                l0_m + (size_t)l * H2, l0_v + (size_t)l * H2, stream);
    launch_gemm(2, hb, w2l0b + (size_t)l * HDIM * H2, y, nullptr,
                N_NODES, H2, HDIM, nullptr, nullptr, nullptr, nullptr, stream);
  }
  leaky_kernel<<<ewBlock, 256, 0, stream>>>(y, x1, n4);

  // ---------------- layer 1 (sf = df = x1; y accumulates into d_out) -------
  float* y1 = (float*)d_out;
  hipMemsetAsync(y1, 0, NH * sizeof(float), stream);
  for (int l = 0; l < 2; ++l) {
    hipMemsetAsync(mx,  0, NH * sizeof(float), stream);
    hipMemsetAsync(den, 0, NH * sizeof(float), stream);
    hipMemsetAsync(num, 0, NH * sizeof(float), stream);
    const int*   srcL = srcAll + offs[l];
    const int*   dstL = dstAll + offs[l];
    const float* eaL  = ea + offs[l];
    const float* WeL  = l1_edge + (size_t)l * HDIM;
    const int eGrid = (lens[l] + 7) / 8;
    edge_max_kernel<<<eGrid, 256, 0, stream>>>(x1, srcL, dstL, eaL, WeL, (int*)mx, lens[l]);
    edge_sum_kernel<<<eGrid, 256, 0, stream>>>(x1, srcL, dstL, eaL, WeL, mx, den, num, lens[l]);
    combine_kernel<<<ewBlock, 256, 0, stream>>>(num, den, x1, outb, n4);
    launch_gemm(1, outb, w1l1b + (size_t)l * H2 * HDIM, nullptr, hb,
                N_NODES, HDIM, H2,
                l1_g + (size_t)l * H2, l1_b + (size_t)l * H2,
                l1_m + (size_t)l * H2, l1_v + (size_t)l * H2, stream);
    launch_gemm(2, hb, w2l1b + (size_t)l * HDIM * H2, y1, nullptr,
                N_NODES, H2, HDIM, nullptr, nullptr, nullptr, nullptr, stream);
  }
}
